// SSMLayer_17102559773344
// MI455X (gfx1250) — matrix operand order repaired and checked by reading, not compiled, not run on hardware
//
#include <hip/hip_runtime.h>
#include <math.h>

// ---------------------------------------------------------------- constants
#define BATCH   16
#define SEQ     2048
#define DM      768          // d_model (= K of the GEMMs)
#define NB      256          // quaternion blocks (DM/3)
#define MROWS   (BATCH*SEQ)  // 32768 GEMM rows
#define NTOT    (2*DM)       // biv (768) ++ inj (768) columns
#define LN_EPS  1e-5f
#define NCHAIN  4096         // BATCH * NB independent scan chains
#define CHUNK   32           // timesteps per scan chunk
#define NCHUNK  (SEQ/CHUNK)  // 64 chunks per chain

typedef __bf16  bf16_t;
typedef bf16_t  v8bf  __attribute__((ext_vector_type(8)));
typedef bf16_t  v16bf __attribute__((ext_vector_type(16)));
typedef float   v8f   __attribute__((ext_vector_type(8)));

// ---------------------------------------------------------------- fp32 -> bf16 hi/lo split
__global__ __launch_bounds__(256) void cvt_split_kernel(const float* __restrict__ in,
                                                        bf16_t* __restrict__ hi,
                                                        bf16_t* __restrict__ lo,
                                                        int n) {
    int i = blockIdx.x * 256 + threadIdx.x;
    if (i < n) {
        float v = in[i];
        bf16_t h = (bf16_t)v;
        hi[i] = h;
        lo[i] = (bf16_t)(v - (float)h);
    }
}

// ---------------------------------------------------------------- WMMA fragment loader
__device__ __forceinline__ v16bf ld_frag(const bf16_t* p0, const bf16_t* p1) {
    v8bf a = *(const v8bf*)p0;
    v8bf b = *(const v8bf*)p1;
    v16bf r;
#pragma unroll
    for (int i = 0; i < 8; ++i) { r[i] = a[i]; r[i + 8] = b[i]; }
    return r;
}

// ---------------------------------------------------------------- compensated bf16 WMMA GEMM
// C[M=32768, N=1536] = Xh*Wh + Xh*Wl + Xl*Wh   (+ bias, split-routed to biv / inj)
__global__ __launch_bounds__(256) void gemm_wmma_kernel(const bf16_t* __restrict__ Ah,
                                                        const bf16_t* __restrict__ Al,
                                                        const bf16_t* __restrict__ Bh,
                                                        const bf16_t* __restrict__ Bl,
                                                        const float* __restrict__ bias_biv,
                                                        const float* __restrict__ bias_in,
                                                        float* __restrict__ bivOut,
                                                        float* __restrict__ injOut) {
    const int lane = threadIdx.x & 31;
    const int wave = threadIdx.x >> 5;
    const int wm = wave & 3;
    const int wn = wave >> 2;
    const int mBase = blockIdx.x * 128 + wm * 32;
    const int nBase = blockIdx.y * 128 + wn * 64;

    const int rowA  = mBase + (lane & 15);
    const int kOffA = (lane >> 4) * 8;
    const int colB  = nBase + (lane & 15);
    const int kOffB = (lane >> 4) * 8;

    v8f acc[2][4] = {};

    for (int kt = 0; kt < DM; kt += 32) {
        v16bf ah[2], al[2], bh[4], bl[4];
#pragma unroll
        for (int mi = 0; mi < 2; ++mi) {
            const size_t o = (size_t)(rowA + mi * 16) * DM + kt + kOffA;
            ah[mi] = ld_frag(Ah + o, Ah + o + 16);
            al[mi] = ld_frag(Al + o, Al + o + 16);
        }
#pragma unroll
        for (int ni = 0; ni < 4; ++ni) {
            const size_t o = (size_t)(colB + ni * 16) * DM + kt + kOffB;
            bh[ni] = ld_frag(Bh + o, Bh + o + 16);
            bl[ni] = ld_frag(Bl + o, Bl + o + 16);
        }
#pragma unroll
        for (int mi = 0; mi < 2; ++mi)
#pragma unroll
            for (int ni = 0; ni < 4; ++ni) {
                acc[mi][ni] = __builtin_amdgcn_wmma_f32_16x16x32_bf16(
                    false, ah[mi], false, bh[ni], (short)0, acc[mi][ni], false, false);
                acc[mi][ni] = __builtin_amdgcn_wmma_f32_16x16x32_bf16(
                    false, ah[mi], false, bl[ni], (short)0, acc[mi][ni], false, false);
                acc[mi][ni] = __builtin_amdgcn_wmma_f32_16x16x32_bf16(
                    false, al[mi], false, bh[ni], (short)0, acc[mi][ni], false, false);
            }
    }

    const int rowC0 = mBase + (lane >> 4) * 8;
    const int colC0 = (lane & 15);
#pragma unroll
    for (int mi = 0; mi < 2; ++mi)
#pragma unroll
        for (int ni = 0; ni < 4; ++ni) {
            const int col = nBase + ni * 16 + colC0;
#pragma unroll
            for (int r = 0; r < 8; ++r) {
                const size_t row = (size_t)(rowC0 + mi * 16 + r);
                float v = acc[mi][ni][r];
                if (col < DM) bivOut[row * DM + col]        = v + bias_biv[col];
                else          injOut[row * DM + (col - DM)] = v + bias_in[col - DM];
            }
        }
}

// ---------------------------------------------------------------- decay GEMV + sigmoid
__global__ __launch_bounds__(256) void decay_kernel(const float* __restrict__ x,
                                                    const float* __restrict__ Wd,
                                                    const float* __restrict__ bd,
                                                    float* __restrict__ out) {
    const int lane = threadIdx.x & 31;
    const int wave = threadIdx.x >> 5;
    const int m = blockIdx.x * 8 + wave;
    const float* xr = x + (size_t)m * DM;
    float s = 0.f;
#pragma unroll
    for (int i = 0; i < DM / 32; ++i)
        s += xr[lane + 32 * i] * Wd[lane + 32 * i];
#pragma unroll
    for (int off = 16; off > 0; off >>= 1)
        s += __shfl_xor(s, off, 32);
    if (lane == 0)
        out[m] = 1.0f / (1.0f + expf(-(s + bd[0])));
}

// ---------------------------------------------------------------- quaternion helpers
// Per-step op from biv (reversed component mapping per reference). Small arguments
// (||biv block|| ~ O(1)) => native v_sin/v_cos are plenty; clamp norm like the ref.
__device__ __forceinline__ void quat_from_biv(float bx, float by, float bz,
                                              float& qw, float& qx, float& qy, float& qz) {
    float nrm = fmaxf(sqrtf(bx * bx + by * by + bz * bz), 1e-8f);
    const float half = 0.5f * nrm;
    qw = __cosf(half);
    const float s = __sinf(half) * __builtin_amdgcn_rcpf(nrm);
    qx = s * bz; qy = -s * by; qz = s * bx;
}

__device__ __forceinline__ void quat_rot(float qw, float qx, float qy, float qz,
                                         float vx, float vy, float vz,
                                         float& rx, float& ry, float& rz) {
    const float tx = 2.f * (qy * vz - qz * vy);
    const float ty = 2.f * (qz * vx - qx * vz);
    const float tz = 2.f * (qx * vy - qy * vx);
    rx = vx + qw * tx + (qy * tz - qz * ty);
    ry = vy + qw * ty + (qz * tx - qx * tz);
    rz = vz + qw * tz + (qx * ty - qy * tx);
}

// ---------------------------------------------------------------- chunked associative scan
// Step op: h -> d*R(q)*h + inj is affine with a scaled-rotation linear part, so ops
// compose: (q2,a2,v2)∘(q1,a1,v1) = (q2⊗q1, a2*a1, a2*R(q2)v1 + v2).
// Pass 1: each (chain, chunk) composes its 32 steps -> one op.  262144 work-items.
__global__ __launch_bounds__(256) void scan_chunk_build(const float* __restrict__ biv,
                                                        const float* __restrict__ inj,
                                                        const float* __restrict__ decay,
                                                        float* __restrict__ chunkOp) {
    const int id = blockIdx.x * 256 + threadIdx.x;      // 0..262143
    const int chain = id & (NCHAIN - 1);                // adjacent lanes -> adjacent chains
    const int chunk = id >> 12;                         // 0..63
    const int b = chain >> 8, j = chain & 255;
    const int t0 = chunk * CHUNK;
    const size_t base = ((size_t)b * SEQ + t0) * DM + (size_t)j * 3;

    float qw = 1.f, qx = 0.f, qy = 0.f, qz = 0.f, a = 1.f;
    float vx = 0.f, vy = 0.f, vz = 0.f;
    for (int i = 0; i < CHUNK; ++i) {
        const size_t o = base + (size_t)i * DM;
        if (i + 4 < CHUNK) {
            __builtin_prefetch(biv + o + 4 * DM, 0, 3);
            __builtin_prefetch(inj + o + 4 * DM, 0, 3);
        }
        const float bx = biv[o], by = biv[o + 1], bz = biv[o + 2];
        const float ix = inj[o], iy = inj[o + 1], iz = inj[o + 2];
        const float d = decay[b * SEQ + t0 + i];
        float tw, tx, ty, tz;
        quat_from_biv(bx, by, bz, tw, tx, ty, tz);
        float rx, ry, rz;
        quat_rot(tw, tx, ty, tz, vx, vy, vz, rx, ry, rz);
        vx = d * rx + ix; vy = d * ry + iy; vz = d * rz + iz;
        // q <- q_t ⊗ q  (Hamilton product; applies existing prefix first)
        const float nw = tw * qw - tx * qx - ty * qy - tz * qz;
        const float nx = tw * qx + tx * qw + ty * qz - tz * qy;
        const float ny = tw * qy - tx * qz + ty * qw + tz * qx;
        const float nz = tw * qz + tx * qy - ty * qx + tz * qw;
        qw = nw; qx = nx; qy = ny; qz = nz;
        a *= d;
    }
    float* op = chunkOp + (size_t)(chain * NCHUNK + chunk) * 8;
    op[0] = qw; op[1] = qx; op[2] = qy; op[3] = qz;
    op[4] = a;  op[5] = vx; op[6] = vy; op[7] = vz;
}

// Pass 2: per chain, apply the 64 chunk ops in order; since h0 = 0 the running
// translation IS the chunk-start state. Sequential depth 64, 4096 lanes.
__global__ __launch_bounds__(256) void scan_chunk_prefix(const float* __restrict__ chunkOp,
                                                         float* __restrict__ hStart) {
    const int chain = blockIdx.x * 256 + threadIdx.x;   // 0..4095
    float vx = 0.f, vy = 0.f, vz = 0.f;
    for (int c = 0; c < NCHUNK; ++c) {
        float* hs = hStart + (size_t)(chain * NCHUNK + c) * 3;
        hs[0] = vx; hs[1] = vy; hs[2] = vz;
        const float* op = chunkOp + (size_t)(chain * NCHUNK + c) * 8;
        const float qw = op[0], qx = op[1], qy = op[2], qz = op[3], a = op[4];
        float rx, ry, rz;
        quat_rot(qw, qx, qy, qz, vx, vy, vz, rx, ry, rz);
        vx = a * rx + op[5]; vy = a * ry + op[6]; vz = a * rz + op[7];
    }
}

// Pass 3: re-run each chunk's 32 steps from its start state, writing h.  262144 items.
__global__ __launch_bounds__(256) void scan_chunk_apply(const float* __restrict__ biv,
                                                        const float* __restrict__ inj,
                                                        const float* __restrict__ decay,
                                                        const float* __restrict__ hStart,
                                                        float* __restrict__ hOut) {
    const int id = blockIdx.x * 256 + threadIdx.x;
    const int chain = id & (NCHAIN - 1);
    const int chunk = id >> 12;
    const int b = chain >> 8, j = chain & 255;
    const int t0 = chunk * CHUNK;
    const size_t base = ((size_t)b * SEQ + t0) * DM + (size_t)j * 3;

    const float* hs = hStart + (size_t)(chain * NCHUNK + chunk) * 3;
    float hx = hs[0], hy = hs[1], hz = hs[2];
    for (int i = 0; i < CHUNK; ++i) {
        const size_t o = base + (size_t)i * DM;
        if (i + 4 < CHUNK) {
            __builtin_prefetch(biv + o + 4 * DM, 0, 3);
            __builtin_prefetch(inj + o + 4 * DM, 0, 3);
        }
        const float bx = biv[o], by = biv[o + 1], bz = biv[o + 2];
        const float ix = inj[o], iy = inj[o + 1], iz = inj[o + 2];
        const float d = decay[b * SEQ + t0 + i];
        float tw, tx, ty, tz;
        quat_from_biv(bx, by, bz, tw, tx, ty, tz);
        float rx, ry, rz;
        quat_rot(tw, tx, ty, tz, hx, hy, hz, rx, ry, rz);
        hx = d * rx + ix; hy = d * ry + iy; hz = d * rz + iz;
        hOut[o] = hx; hOut[o + 1] = hy; hOut[o + 2] = hz;
    }
}

// ---------------------------------------------------------------- LayerNorm + residual
__global__ __launch_bounds__(256) void ln_kernel(const float* __restrict__ h,
                                                 const float* __restrict__ x,
                                                 const float* __restrict__ gamma,
                                                 const float* __restrict__ beta,
                                                 float* __restrict__ out) {
    const int m = blockIdx.x;
    const int tid = threadIdx.x;
    const int lane = tid & 31, wave = tid >> 5;
    const float* hr = h + (size_t)m * DM;
    const float* xr = x + (size_t)m * DM;
    float* orow = out + (size_t)m * DM;

    float v[3];
    float s1 = 0.f, s2 = 0.f;
#pragma unroll
    for (int k = 0; k < 3; ++k) {
        v[k] = hr[tid + 256 * k];
        s1 += v[k];
        s2 += v[k] * v[k];
    }
#pragma unroll
    for (int off = 16; off > 0; off >>= 1) {
        s1 += __shfl_xor(s1, off, 32);
        s2 += __shfl_xor(s2, off, 32);
    }
    __shared__ float p1[8], p2[8], stats[2];
    if (lane == 0) { p1[wave] = s1; p2[wave] = s2; }
    __syncthreads();
    if (wave == 0) {
        float a = (lane < 8) ? p1[lane] : 0.f;
        float bqs = (lane < 8) ? p2[lane] : 0.f;
#pragma unroll
        for (int off = 4; off > 0; off >>= 1) {
            a += __shfl_xor(a, off, 32);
            bqs += __shfl_xor(bqs, off, 32);
        }
        if (lane == 0) {
            const float mean = a * (1.0f / DM);
            const float var = bqs * (1.0f / DM) - mean * mean;
            stats[0] = mean;
            stats[1] = rsqrtf(var + LN_EPS);
        }
    }
    __syncthreads();
    const float mean = stats[0], rstd = stats[1];
#pragma unroll
    for (int k = 0; k < 3; ++k) {
        const int i = tid + 256 * k;
        orow[i] = (v[k] - mean) * rstd * gamma[i] + beta[i] + xr[i];
    }
}

// ---------------------------------------------------------------- launcher
extern "C" void kernel_launch(void* const* d_in, const int* in_sizes, int n_in,
                              void* d_out, int out_size, void* d_ws, size_t ws_size,
                              hipStream_t stream) {
    const float* x       = (const float*)d_in[0];
    const float* W_biv   = (const float*)d_in[1];
    const float* b_biv   = (const float*)d_in[2];
    const float* W_decay = (const float*)d_in[3];
    const float* b_decay = (const float*)d_in[4];
    const float* W_in    = (const float*)d_in[5];
    const float* b_in    = (const float*)d_in[6];
    const float* ln_g    = (const float*)d_in[7];
    const float* ln_b    = (const float*)d_in[8];
    float* out = (float*)d_out;

    char* ws = (char*)d_ws;
    const size_t SZ_XBF = (size_t)MROWS * DM * sizeof(bf16_t);   // 48 MB each
    const size_t SZ_WBF = (size_t)DM * NTOT * sizeof(bf16_t);    // 2.25 MB each
    const size_t SZ_F32 = (size_t)MROWS * DM * sizeof(float);    // 96 MB each

    // region 0: xh ++ xl during GEMM phase, reused as h afterwards (same 96 MB)
    bf16_t* xh = (bf16_t*)(ws);
    bf16_t* xl = (bf16_t*)(ws + SZ_XBF);
    float*  hb = (float*)(ws);
    size_t o = 2 * SZ_XBF;
    bf16_t* Wh = (bf16_t*)(ws + o); o += SZ_WBF;
    bf16_t* Wl = (bf16_t*)(ws + o); o += SZ_WBF;
    float* bivB = (float*)(ws + o); o += SZ_F32;
    float* injB = (float*)(ws + o); o += SZ_F32;
    float* decB = (float*)(ws + o); o += (size_t)MROWS * sizeof(float);
    float* copB = (float*)(ws + o); o += (size_t)NCHAIN * NCHUNK * 8 * sizeof(float); // 8 MB
    float* hstB = (float*)(ws + o); o += (size_t)NCHAIN * NCHUNK * 3 * sizeof(float); // 3 MB
    (void)ws_size; (void)in_sizes; (void)n_in; (void)out_size;

    // 1) bf16 hi/lo splits of activations and (concatenated, K-contiguous) weights
    {
        const int n = MROWS * DM;
        cvt_split_kernel<<<(n + 255) / 256, 256, 0, stream>>>(x, xh, xl, n);
        const int nw = DM * DM;
        cvt_split_kernel<<<(nw + 255) / 256, 256, 0, stream>>>(W_biv, Wh, Wl, nw);
        cvt_split_kernel<<<(nw + 255) / 256, 256, 0, stream>>>(W_in, Wh + (size_t)DM * DM,
                                                               Wl + (size_t)DM * DM, nw);
    }
    // 2) decay GEMV (full fp32, cheap)
    decay_kernel<<<MROWS / 8, 256, 0, stream>>>(x, W_decay, b_decay, decB);
    // 3) compensated bf16 WMMA GEMM -> biv, inj
    {
        dim3 grid(MROWS / 128, NTOT / 128);
        gemm_wmma_kernel<<<grid, 256, 0, stream>>>(xh, xl, Wh, Wl, b_biv, b_in, bivB, injB);
    }
    // 4) chunked associative scan: depth 2048 -> 32 + 64 + 32
    scan_chunk_build<<<(NCHAIN * NCHUNK) / 256, 256, 0, stream>>>(bivB, injB, decB, copB);
    scan_chunk_prefix<<<NCHAIN / 256, 256, 0, stream>>>(copB, hstB);
    scan_chunk_apply<<<(NCHAIN * NCHUNK) / 256, 256, 0, stream>>>(bivB, injB, decB, hstB, hb);
    // 5) LayerNorm + residual
    ln_kernel<<<MROWS, 256, 0, stream>>>(hb, x, ln_g, ln_b, out);
}
